// VQVAE_6081673691352
// MI455X (gfx1250) — compile-verified, matrix operand order checked by reading, not run on hardware
//
#include <hip/hip_runtime.h>
#include <hip/hip_bf16.h>

typedef __bf16 bf16;
typedef __attribute__((ext_vector_type(16))) __bf16 v16bf;
typedef __attribute__((ext_vector_type(8)))  float  v8f;

__device__ __forceinline__ bf16 f2bf(float f) {
    unsigned u = __float_as_uint(f);
    u += 0x7fffu + ((u >> 16) & 1u);
    unsigned short h = (unsigned short)(u >> 16);
    bf16 r;
    __builtin_memcpy(&r, &h, 2);
    return r;
}
__device__ __forceinline__ float bf2f(bf16 b) {
    unsigned short h;
    __builtin_memcpy(&h, &b, 2);
    return __uint_as_float(((unsigned)h) << 16);
}

// ---------------------------------------------------------------------------
// WMMA macro-tile: 8 waves, block tile 64(M) x 64(N) x 64(K).
// Wave (w&3) -> M quadrant, (w>>2) -> N half; each wave owns 16x32 C.
// 4 v_wmma_f32_16x16x32_bf16 per wave per barrier interval.
// ---------------------------------------------------------------------------
__device__ __forceinline__ void wmma_block(const bf16 (*As)[72], const bf16 (*Bs)[72],
                                           int mo, int no, int lane,
                                           v8f& acc0, v8f& acc1) {
    const int r = lane & 15, kb = (lane >> 4) * 8;
#pragma unroll
    for (int kh = 0; kh < 2; ++kh) {
        v16bf a, b0, b1;
#pragma unroll
        for (int v = 0; v < 8; ++v) {
            const int k0 = kh * 32 + kb + ((v & 3) << 1) + ((v >> 2) << 4);
            a[2 * v]      = As[mo + r][k0];
            a[2 * v + 1]  = As[mo + r][k0 + 1];
            b0[2 * v]     = Bs[no + r][k0];
            b0[2 * v + 1] = Bs[no + r][k0 + 1];
            b1[2 * v]     = Bs[no + 16 + r][k0];
            b1[2 * v + 1] = Bs[no + 16 + r][k0 + 1];
        }
        acc0 = __builtin_amdgcn_wmma_f32_16x16x32_bf16(false, a, false, b0,
                                                       (short)0, acc0, false, false);
        acc1 = __builtin_amdgcn_wmma_f32_16x16x32_bf16(false, a, false, b1,
                                                       (short)0, acc1, false, false);
    }
}

// ---------------------------------------------------------------------------
// Prep kernels
// ---------------------------------------------------------------------------
__global__ void zero16_kernel(uint4* __restrict__ dst, int n16) {
    int i = blockIdx.x * 256 + threadIdx.x;
    if (i < n16) dst[i] = make_uint4(0, 0, 0, 0);
}

__global__ void to_bf16_kernel(const float* __restrict__ src, bf16* __restrict__ dst, int n) {
    int i = blockIdx.x * 256 + threadIdx.x;
    if (i < n) dst[i] = f2bf(src[i]);
}

__global__ void cb_prep_kernel(const float* __restrict__ cb, bf16* __restrict__ cbb,
                               float* __restrict__ cbn) {
    const int row = blockIdx.x, tid = threadIdx.x;
    float s = 0.f;
    for (int d = tid; d < 2048; d += 256) {
        float v = cb[row * 2048 + d];
        cbb[row * 2048 + d] = f2bf(v);
        s += v * v;
    }
    __shared__ float red[256];
    red[tid] = s; __syncthreads();
    for (int st = 128; st > 0; st >>= 1) {
        if (tid < st) red[tid] += red[tid + st];
        __syncthreads();
    }
    if (tid == 0) cbn[row] = red[0];
}

// ConvTranspose2d prepack: [Cin][Cout][4][4] f32 -> 4 parity classes of [Cout][Cin*4] bf16.
__global__ void deconv_prepack_kernel(const float* __restrict__ w, bf16* __restrict__ out,
                                      int Cin, int Cout) {
    const int total = 4 * Cout * Cin * 4;
    int gid = blockIdx.x * 256 + threadIdx.x;
    if (gid >= total) return;
    const int perClass = Cout * Cin * 4;
    int cls = gid / perClass, r = gid % perClass;
    int co = r / (Cin * 4), r2 = r % (Cin * 4);
    int ci = r2 >> 2, t = r2 & 3;
    int ey = cls >> 1, ex = cls & 1, tky = t >> 1, tkx = t & 1;
    int ky = ey ? (tky ? 2 : 0) : (tky ? 3 : 1);
    int kx = ex ? (tkx ? 2 : 0) : (tkx ? 3 : 1);
    out[gid] = f2bf(w[((size_t)ci * Cout + co) * 16 + ky * 4 + kx]);
}

__global__ void init_kernel(unsigned long long* __restrict__ best, float* __restrict__ loss) {
    int gid = blockIdx.x * 256 + threadIdx.x;
    if (gid < 8192) best[gid] = ~0ull;
    if (gid == 0) *loss = 0.f;
}

// ---------------------------------------------------------------------------
// conv1: 3->128, k=4, s=2, p=1, ReLU. Direct. Output padded [8][128][130][130].
// ---------------------------------------------------------------------------
__global__ __launch_bounds__(256) void conv1_kernel(const float* __restrict__ x,
                                                    const float* __restrict__ w,
                                                    const float* __restrict__ bias,
                                                    bf16* __restrict__ out) {
    __shared__ float ws[128 * 48];
    for (int i = threadIdx.x; i < 128 * 48; i += 256) ws[i] = w[i];
    __syncthreads();
    const int gid = blockIdx.x * 256 + threadIdx.x;   // over 8*128*128
    const int b = gid >> 14, rem = gid & 16383, oy = rem >> 7, ox = rem & 127;
    float xin[48];
#pragma unroll
    for (int ci = 0; ci < 3; ++ci)
#pragma unroll
        for (int ky = 0; ky < 4; ++ky)
#pragma unroll
            for (int kx = 0; kx < 4; ++kx) {
                int iy = 2 * oy - 1 + ky, ix = 2 * ox - 1 + kx;
                float v = 0.f;
                if ((unsigned)iy < 256u && (unsigned)ix < 256u)
                    v = x[((size_t)(b * 3 + ci) * 256 + iy) * 256 + ix];
                xin[ci * 16 + ky * 4 + kx] = v;
            }
    for (int co = 0; co < 128; ++co) {
        float acc = bias[co];
        const float* wr = &ws[co * 48];
#pragma unroll
        for (int k = 0; k < 48; ++k) acc += xin[k] * wr[k];
        acc = fmaxf(acc, 0.f);
        out[((size_t)(b * 128 + co) * 130 + oy + 1) * 130 + ox + 1] = f2bf(acc);
    }
}

// ---------------------------------------------------------------------------
// Implicit-GEMM conv (k=4,s=2,p=1). Input is zero-padded NCHW: [B][Cin][H+2][W+2]
// (logical dims H,W passed). M=Cout, N=B*OH*OW, K=Cin*16, all tiles exact.
// Gather is UNCONDITIONAL: 8x 4-byte loads per thread per K-step (provably
// 4B-aligned: window origin at even column, row pitch (W+2)*2 multiple of 4).
// tposed: bf16 out as [pixel][Cout]; padout: bf16 out padded NCHW (+1 halo).
// ---------------------------------------------------------------------------
__global__ __launch_bounds__(256) void conv_gemm_kernel(
    const bf16* __restrict__ in, const bf16* __restrict__ w, const float* __restrict__ bias,
    bf16* __restrict__ outb, float* __restrict__ outf,
    int B, int Cin, int H, int W, int Cout, int OH, int OW,
    int relu, int tposed, int padout) {
    __shared__ __align__(16) bf16 As[64][72];
    __shared__ __align__(16) bf16 Bs[64][72];
    const int tid = threadIdx.x, lane = tid & 31, wave = tid >> 5;
    const int m0 = blockIdx.x * 64, n0 = blockIdx.y * 64;
    const int K = Cin * 16, HW = OH * OW;
    const int WP = W + 2;
    const size_t plane = (size_t)(H + 2) * WP;
    const int mo = (wave & 3) * 16, no = (wave >> 2) * 32;
    const int ml = tid >> 2, kseg = (tid & 3) * 16;
    // B-gather window origin for this thread's n row (fixed across K loop)
    const int bn = n0 + ml;
    const int bb = bn / HW, brem = bn % HW, boy = brem / OW, box = brem % OW;
    const bf16* wnd0 = in + (size_t)bb * Cin * plane + (size_t)(2 * boy) * WP + 2 * box;
    v8f acc0 = {}, acc1 = {};
    for (int kk = 0; kk < K; kk += 64) {
        if (kk + 64 < K)
            __builtin_prefetch(&w[(size_t)(m0 + ml) * K + kk + 64], 0, 1);
        {   // A: weights, contiguous 16B vector copies
            const bf16* src = &w[(size_t)(m0 + ml) * K + kk + kseg];
            *(uint4*)&As[ml][kseg]     = *(const uint4*)src;
            *(uint4*)&As[ml][kseg + 8] = *(const uint4*)(src + 8);
        }
        {   // B: one full 4x4 padded window of one ci, unconditional b32 pairs
            const bf16* wnd = wnd0 + (size_t)((kk + kseg) >> 4) * plane;
            unsigned tw[8];
#pragma unroll
            for (int ky = 0; ky < 4; ++ky) {
                tw[ky * 2]     = *(const unsigned*)(wnd + ky * WP);
                tw[ky * 2 + 1] = *(const unsigned*)(wnd + ky * WP + 2);
            }
            __builtin_memcpy(&Bs[ml][kseg], tw, 32);
        }
        __syncthreads();
        wmma_block(As, Bs, mo, no, lane, acc0, acc1);
        __syncthreads();
    }
#pragma unroll
    for (int f = 0; f < 2; ++f) {
        v8f& a = f ? acc1 : acc0;
#pragma unroll
        for (int r = 0; r < 8; ++r) {
            int m = m0 + mo + r + ((lane >> 4) << 3);
            int n = n0 + no + f * 16 + (lane & 15);
            float v = a[r] + bias[m];
            if (relu) v = fmaxf(v, 0.f);
            int b = n / HW, rem = n % HW, oy = rem / OW, ox = rem % OW;
            size_t idx;
            if (tposed)      idx = (size_t)(b * HW + rem) * Cout + m;
            else if (padout) idx = ((size_t)(b * Cout + m) * (OH + 2) + oy + 1) * (OW + 2) + ox + 1;
            else             idx = (size_t)(b * Cout + m) * HW + rem;
            outb[idx] = f2bf(v);
            if (outf) outf[(size_t)(b * Cout + m) * HW + rem] = v;
        }
    }
}

// ---------------------------------------------------------------------------
// VQ distances + fused argmin. z_e bf16 [p=8192][d=2048], codebook bf16 [c][d].
// score[p][c] = ||c||^2 - 2 z_p.c ; key = orderMap(score)<<32 | c -> atomicMin.
// ---------------------------------------------------------------------------
__global__ __launch_bounds__(256) void dist_wmma_kernel(
    const bf16* __restrict__ zet, const bf16* __restrict__ cbb,
    const float* __restrict__ cbn, unsigned long long* __restrict__ best) {
    __shared__ __align__(16) bf16 As[64][72];
    __shared__ __align__(16) bf16 Bs[64][72];
    const int tid = threadIdx.x, lane = tid & 31, wave = tid >> 5;
    const int p0 = blockIdx.x * 64, c0 = blockIdx.y * 64;
    const int mo = (wave & 3) * 16, no = (wave >> 2) * 32;
    const int ml = tid >> 2, kseg = (tid & 3) * 16;
    v8f acc0 = {}, acc1 = {};
    for (int kk = 0; kk < 2048; kk += 64) {
        if (kk + 64 < 2048) {
            __builtin_prefetch(&cbb[(size_t)(c0 + ml) * 2048 + kk + 64], 0, 1);
            __builtin_prefetch(&zet[(size_t)(p0 + ml) * 2048 + kk + 64], 0, 1);
        }
        const bf16* sa = &zet[(size_t)(p0 + ml) * 2048 + kk + kseg];
        *(uint4*)&As[ml][kseg]     = *(const uint4*)sa;
        *(uint4*)&As[ml][kseg + 8] = *(const uint4*)(sa + 8);
        const bf16* sb = &cbb[(size_t)(c0 + ml) * 2048 + kk + kseg];
        *(uint4*)&Bs[ml][kseg]     = *(const uint4*)sb;
        *(uint4*)&Bs[ml][kseg + 8] = *(const uint4*)(sb + 8);
        __syncthreads();
        wmma_block(As, Bs, mo, no, lane, acc0, acc1);
        __syncthreads();
    }
#pragma unroll
    for (int f = 0; f < 2; ++f) {
        v8f& a = f ? acc1 : acc0;
#pragma unroll
        for (int r = 0; r < 8; ++r) {
            int p = p0 + mo + r + ((lane >> 4) << 3);
            int c = c0 + no + f * 16 + (lane & 15);
            float s = cbn[c] - 2.f * a[r];
            unsigned fu = __float_as_uint(s);
            unsigned mk = (fu & 0x80000000u) ? ~fu : (fu | 0x80000000u);
            unsigned long long key = ((unsigned long long)mk << 32) | (unsigned)c;
            atomicMin(&best[p], key);
        }
    }
}

// ---------------------------------------------------------------------------
// Gather z_q (f32 codebook) -> padded bf16 NCHW [8][2048][34][34]; loss num.
// ---------------------------------------------------------------------------
__global__ __launch_bounds__(256) void quant_gather_kernel(
    const unsigned long long* __restrict__ best, const float* __restrict__ cb,
    const float* __restrict__ zef, bf16* __restrict__ zqp, float* __restrict__ loss) {
    const int p = blockIdx.x, tid = threadIdx.x;
    const int b = p >> 10, hw = p & 1023, y = hw >> 5, xq = hw & 31;
    const unsigned idx = (unsigned)(best[p] & 0xFFFFFFFFull);
    float local = 0.f;
    for (int d = tid; d < 2048; d += 256) {
        float zq = cb[(size_t)idx * 2048 + d];
        float diff = zef[(size_t)(b * 2048 + d) * 1024 + hw] - zq;
        local += diff * diff;
        zqp[((size_t)(b * 2048 + d) * 34 + y + 1) * 34 + xq + 1] = f2bf(zq);
    }
    __shared__ float red[256];
    red[tid] = local; __syncthreads();
    for (int st = 128; st > 0; st >>= 1) {
        if (tid < st) red[tid] += red[tid + st];
        __syncthreads();
    }
    if (tid == 0) atomicAdd(loss, red[0]);
}

// ---------------------------------------------------------------------------
// Implicit-GEMM ConvTranspose2d (k=4,s=2,p=1) per output-parity class.
// Input zero-padded NCHW [B][Cin][IH+2][IW+2]. Gather is 16 unconditional u16
// loads (tap offsets {0,-1,-IWP,-IWP-1} from per-pixel origin).
// ---------------------------------------------------------------------------
__global__ __launch_bounds__(256) void deconv_gemm_kernel(
    const bf16* __restrict__ in, const bf16* __restrict__ wpk, const float* __restrict__ bias,
    bf16* __restrict__ out, int B, int Cin, int IH, int IW, int Cout,
    int relu, int padout) {
    __shared__ __align__(16) bf16 As[64][72];
    __shared__ __align__(16) bf16 Bs[64][72];
    const int tid = threadIdx.x, lane = tid & 31, wave = tid >> 5;
    const int cls = blockIdx.z, ey = cls >> 1, ex = cls & 1;
    const int m0 = blockIdx.x * 64, n0 = blockIdx.y * 64;
    const int K = Cin * 4, HW = IH * IW;
    const int OH = IH * 2, OW = IW * 2;
    const int IWP = IW + 2;
    const size_t plane = (size_t)(IH + 2) * IWP;
    const bf16* wc = wpk + (size_t)cls * Cout * K;
    const int mo = (wave & 3) * 16, no = (wave >> 2) * 32;
    const int ml = tid >> 2, kseg = (tid & 3) * 16;
    const int bn = n0 + ml;
    const int bb = bn / HW, brem = bn % HW, boy2 = brem / IW, box2 = brem % IW;
    // per-pixel tap origin in padded coords: row = oy2+ey+1, col = ox2+ex+1
    const bf16* org0 = in + (size_t)bb * Cin * plane
                          + (size_t)(boy2 + ey + 1) * IWP + (box2 + ex + 1);
    v8f acc0 = {}, acc1 = {};
    for (int kk = 0; kk < K; kk += 64) {
        if (kk + 64 < K)
            __builtin_prefetch(&wc[(size_t)(m0 + ml) * K + kk + 64], 0, 1);
        {
            const bf16* src = &wc[(size_t)(m0 + ml) * K + kk + kseg];
            *(uint4*)&As[ml][kseg]     = *(const uint4*)src;
            *(uint4*)&As[ml][kseg + 8] = *(const uint4*)(src + 8);
        }
        {   // gather: 4 ci's x 4 taps, unconditional
            const int ci0 = (kk + kseg) >> 2;
            bf16 tmp[16];
#pragma unroll
            for (int jc = 0; jc < 4; ++jc) {
                const bf16* bp = org0 + (size_t)(ci0 + jc) * plane;
                tmp[jc * 4 + 0] = bp[0];
                tmp[jc * 4 + 1] = bp[-1];
                tmp[jc * 4 + 2] = bp[-IWP];
                tmp[jc * 4 + 3] = bp[-IWP - 1];
            }
            __builtin_memcpy(&Bs[ml][kseg], tmp, 32);
        }
        __syncthreads();
        wmma_block(As, Bs, mo, no, lane, acc0, acc1);
        __syncthreads();
    }
#pragma unroll
    for (int f = 0; f < 2; ++f) {
        v8f& a = f ? acc1 : acc0;
#pragma unroll
        for (int r = 0; r < 8; ++r) {
            int m = m0 + mo + r + ((lane >> 4) << 3);
            int n = n0 + no + f * 16 + (lane & 15);
            float v = a[r] + bias[m];
            if (relu) v = fmaxf(v, 0.f);
            int b = n / HW, rem = n % HW, oy2 = rem / IW, ox2 = rem % IW;
            int oy = 2 * oy2 + ey, ox = 2 * ox2 + ex;
            size_t idx = padout
                ? (((size_t)(b * Cout + m) * (OH + 2) + oy + 1) * (OW + 2) + ox + 1)
                : (((size_t)(b * Cout + m) * OH + oy) * OW + ox);
            out[idx] = f2bf(v);
        }
    }
}

// ---------------------------------------------------------------------------
// dec3: ConvTranspose2d 128->1 + sigmoid (M=1: direct, memory-bound).
// ---------------------------------------------------------------------------
__global__ __launch_bounds__(256) void dec3_kernel(const bf16* __restrict__ in,
                                                   const float* __restrict__ w,
                                                   const float* __restrict__ b3,
                                                   float* __restrict__ out) {
    __shared__ float ws[2048];
    for (int i = threadIdx.x; i < 2048; i += 256) ws[i] = w[i];
    __syncthreads();
    const int gid = blockIdx.x * 256 + threadIdx.x;  // over 8*256*256
    const int b = gid >> 16, rem = gid & 65535, oy = rem >> 8, ox = rem & 255;
    const int ey = oy & 1, ex = ox & 1;
    float acc = b3[0];
#pragma unroll
    for (int tky = 0; tky < 2; ++tky) {
        int ky = ey ? (tky ? 2 : 0) : (tky ? 3 : 1);
        int iy = (oy + 1 - ky) >> 1;
        if ((unsigned)iy >= 128u) continue;
#pragma unroll
        for (int tkx = 0; tkx < 2; ++tkx) {
            int kx = ex ? (tkx ? 2 : 0) : (tkx ? 3 : 1);
            int ix = (ox + 1 - kx) >> 1;
            if ((unsigned)ix >= 128u) continue;
            const bf16* ip = in + (size_t)b * 128 * 16384 + (size_t)iy * 128 + ix;
            for (int ci = 0; ci < 128; ++ci)
                acc += bf2f(ip[(size_t)ci * 16384]) * ws[ci * 16 + ky * 4 + kx];
        }
    }
    out[gid] = 1.f / (1.f + __expf(-acc));
}

__global__ void loss_final_kernel(const float* __restrict__ loss, float* __restrict__ out) {
    if (blockIdx.x == 0 && threadIdx.x == 0)
        *out = *loss * (1.25f / 16777216.0f);   // 1.25 * mean over 8*2048*32*32
}

// ---------------------------------------------------------------------------
extern "C" void kernel_launch(void* const* d_in, const int* in_sizes, int n_in,
                              void* d_out, int out_size, void* d_ws, size_t ws_size,
                              hipStream_t stream) {
    const float* x        = (const float*)d_in[0];
    const float* enc_w1   = (const float*)d_in[1];
    const float* enc_b1   = (const float*)d_in[2];
    const float* enc_w2   = (const float*)d_in[3];
    const float* enc_b2   = (const float*)d_in[4];
    const float* enc_w3   = (const float*)d_in[5];
    const float* enc_b3   = (const float*)d_in[6];
    const float* codebook = (const float*)d_in[7];
    const float* dec_w1   = (const float*)d_in[8];
    const float* dec_b1   = (const float*)d_in[9];
    const float* dec_w2   = (const float*)d_in[10];
    const float* dec_b2   = (const float*)d_in[11];
    const float* dec_w3   = (const float*)d_in[12];
    const float* dec_b3   = (const float*)d_in[13];

    char* p = (char*)d_ws;
    auto take = [&](size_t bytes) {
        char* r = p;
        p += (bytes + 255) & ~(size_t)255;
        return r;
    };
    const size_t H1P = 17305600ull;   // 8*128*130*130
    const size_t H2P = 4460544ull;    // 8*128*66*66
    const size_t ZQP = 18939904ull;   // 8*2048*34*34
    bf16*  h1p  = (bf16*)take(H1P * 2);             // padded [8,128,130,130]
    bf16*  h2p  = (bf16*)take(H2P * 2);             // padded [8,128,66,66]
    float* zef  = (float*)take(16777216ull * 4);    // z_e f32 NCHW [8,2048,32,32]
    bf16*  zet  = (bf16*)take(16777216ull * 2);     // z_e bf16 transposed [8192][2048]
    bf16*  w2b  = (bf16*)take(262144ull * 2);
    bf16*  w3b  = (bf16*)take(4194304ull * 2);
    bf16*  cbb  = (bf16*)take(2097152ull * 2);
    float* cbn  = (float*)take(1024ull * 4);
    bf16*  dw1p = (bf16*)take(4194304ull * 2);      // 4 classes x [128][8192]
    bf16*  dw2p = (bf16*)take(262144ull * 2);       // 4 classes x [128][512]
    unsigned long long* best = (unsigned long long*)take(8192ull * 8);
    bf16*  zqp  = (bf16*)take(ZQP * 2);             // padded z_q [8,2048,34,34]
    bf16*  d1p  = (bf16*)take(H2P * 2);             // padded [8,128,66,66]
    bf16*  d2   = (bf16*)take(16777216ull * 2);     // [8,128,128,128]
    float* loss = (float*)take(256);

    float* out = (float*)d_out;                      // [524288 recon | 1 loss]

    // --- zero-fill padded activation buffers (halo must be 0 every call) ---
    auto zfill = [&](void* ptr, size_t bytes) {
        int n16 = (int)(bytes / 16);
        zero16_kernel<<<(n16 + 255) / 256, 256, 0, stream>>>((uint4*)ptr, n16);
    };
    zfill(h1p, H1P * 2);
    zfill(h2p, H2P * 2);
    zfill(zqp, ZQP * 2);
    zfill(d1p, H2P * 2);

    // --- weight prep ---
    to_bf16_kernel<<<(262144 + 255) / 256, 256, 0, stream>>>(enc_w2, w2b, 262144);
    to_bf16_kernel<<<(4194304 + 255) / 256, 256, 0, stream>>>(enc_w3, w3b, 4194304);
    cb_prep_kernel<<<1024, 256, 0, stream>>>(codebook, cbb, cbn);
    deconv_prepack_kernel<<<(4194304 + 255) / 256, 256, 0, stream>>>(dec_w1, dw1p, 2048, 128);
    deconv_prepack_kernel<<<(262144 + 255) / 256, 256, 0, stream>>>(dec_w2, dw2p, 128, 128);
    init_kernel<<<32, 256, 0, stream>>>(best, loss);

    // --- encoder ---
    conv1_kernel<<<512, 256, 0, stream>>>(x, enc_w1, enc_b1, h1p);
    conv_gemm_kernel<<<dim3(2, 512), 256, 0, stream>>>(h1p, w2b, enc_b2, h2p, nullptr,
                                                       8, 128, 128, 128, 128, 64, 64,
                                                       1, 0, 1);
    conv_gemm_kernel<<<dim3(32, 128), 256, 0, stream>>>(h2p, w3b, enc_b3, zet, zef,
                                                        8, 128, 64, 64, 2048, 32, 32,
                                                        0, 1, 0);

    // --- vector quantization ---
    dist_wmma_kernel<<<dim3(128, 16), 256, 0, stream>>>(zet, cbb, cbn, best);
    quant_gather_kernel<<<8192, 256, 0, stream>>>(best, codebook, zef, zqp, loss);

    // --- decoder ---
    deconv_gemm_kernel<<<dim3(2, 128, 4), 256, 0, stream>>>(zqp, dw1p, dec_b1, d1p,
                                                            8, 2048, 32, 32, 128, 1, 1);
    deconv_gemm_kernel<<<dim3(2, 512, 4), 256, 0, stream>>>(d1p, dw2p, dec_b2, d2,
                                                            8, 128, 64, 64, 128, 1, 0);
    dec3_kernel<<<2048, 256, 0, stream>>>(d2, dec_w3, dec_b3, out);
    loss_final_kernel<<<1, 1, 0, stream>>>(loss, out + 524288);
}
